// SAR_50818053046739
// MI455X (gfx1250) — matrix operand order repaired and checked by reading, not compiled, not run on hardware
//
#include <hip/hip_runtime.h>
#include <stdint.h>

typedef __attribute__((ext_vector_type(16))) _Float16 v16h;
typedef __attribute__((ext_vector_type(8)))  _Float16 v8h;
typedef __attribute__((ext_vector_type(2)))  _Float16 v2h;
typedef __attribute__((ext_vector_type(8)))  float    v8f;

namespace {
constexpr int kB   = 256;
constexpr int kH   = 8;
constexpr int kW   = 32;
constexpr int kC   = 512;
constexpr int kR   = 512;   // RNN == ATT
constexpr int kV1  = 127;   // vocab + 1
constexpr int kT   = 31;
constexpr int kHW  = kH * kW;        // 256 positions
constexpr int kBHW = kB * kHW;       // 65536 rows for conv-GEMM
constexpr int kKc  = 9 * kC;         // 4608 conv GEMM K
constexpr int kPH  = kH + 2;         // padded height 10
constexpr int kPW  = kW + 2;         // padded width 34
}

__device__ __forceinline__ unsigned lds_off(const void* p) {
  // LDS aperture: low 32 bits of the flat address are the LDS byte offset.
  return (unsigned)(unsigned long long)p;
}

// Issue one double-buffer batch: 4 A sub-tiles (16x32 each) + 1 B^T tile,
// 32B per lane per tile -> 10 async b128 transfers (ASYNCcnt += 10).
// The 24-bit inst offset applies to BOTH LDS and global addresses (ISA 10.x),
// so offset:16 reuses the same address VGPRs for the second 16B.
__device__ __forceinline__ void stage_async10(
    unsigned lA0, unsigned lA1, unsigned lA2, unsigned lA3, unsigned lB,
    const _Float16* a0, const _Float16* a1, const _Float16* a2,
    const _Float16* a3, const _Float16* bp) {
  asm volatile(
      "s_wait_dscnt 0\n\t"  // prior DS reads of this buffer done before overwrite
      "global_load_async_to_lds_b128 %0, %5, off\n\t"
      "global_load_async_to_lds_b128 %0, %5, off offset:16\n\t"
      "global_load_async_to_lds_b128 %1, %6, off\n\t"
      "global_load_async_to_lds_b128 %1, %6, off offset:16\n\t"
      "global_load_async_to_lds_b128 %2, %7, off\n\t"
      "global_load_async_to_lds_b128 %2, %7, off offset:16\n\t"
      "global_load_async_to_lds_b128 %3, %8, off\n\t"
      "global_load_async_to_lds_b128 %3, %8, off offset:16\n\t"
      "global_load_async_to_lds_b128 %4, %9, off\n\t"
      "global_load_async_to_lds_b128 %4, %9, off offset:16"
      :: "v"(lA0), "v"(lA1), "v"(lA2), "v"(lA3), "v"(lB),
         "v"((unsigned long long)a0), "v"((unsigned long long)a1),
         "v"((unsigned long long)a2), "v"((unsigned long long)a3),
         "v"((unsigned long long)bp)
      : "memory");
}
__device__ __forceinline__ void wait_async_prev() {   // previous batch landed
  asm volatile("s_wait_asynccnt 10" ::: "memory");
}
__device__ __forceinline__ void wait_async_all() {
  asm volatile("s_wait_asynccnt 0" ::: "memory");
}

// ---------------------------------------------------------------------------
// WMMA core on one staged buffer: 4 M-tiles x 1 shared B tile -> 4 WMMAs.
// Buffer layout (halves): [A0 512][A1 512][A2 512][A3 512][B^T 512].
// A and B use the same contiguous per-lane k runs (not the ISA table order):
//   A lane half hi: K {hi*16..+15}                 -> one 32B DS load
//   B lane half hi: K {hi*16..+15}                 -> one 32B DS load
// ---------------------------------------------------------------------------
__device__ __forceinline__ void compute4(const _Float16* buf, v8f (&acc)[4],
                                         int lane) {
  const int row = lane & 15;
  const int hi  = lane >> 4;
  const v16h b = *(const v16h*)(buf + 2048 + row * 32 + hi * 16);
#pragma unroll
  for (int ti = 0; ti < 4; ++ti) {
    const _Float16* As = buf + ti * 512;
    const v16h a = *(const v16h*)(As + row * 32 + hi * 16);
    acc[ti] = __builtin_amdgcn_wmma_f32_16x16x32_f16(
        false, a, false, b, (short)0, acc[ti], false, false);
  }
}

// ---------------------------------------------------------------------------
// Generic f16 GEMM: C(MxN) = A(MxK) * B(KxN) + bias, B supplied TRANSPOSED
// (ntN*16 rows x K). M%64==0, K%32==0 and K/32 even. Each wave owns a 64x16
// output block; double-buffered async staging; no block barriers.
// 4 waves/block (128 threads), 40KB LDS.
// ---------------------------------------------------------------------------
__global__ void gemm_wmma_kernel(const _Float16* __restrict__ A, int lda,
                                 const _Float16* __restrict__ BT, int ldbt,
                                 const float* __restrict__ bias,
                                 float* __restrict__ Cf,
                                 _Float16* __restrict__ Ch, int ldc,
                                 int M, int N, int K, int ntN, int ntTotal) {
  __shared__ __align__(64) _Float16 smem[4 * 5120];
  const int wid = threadIdx.x >> 5, lane = threadIdx.x & 31;
  int tile = blockIdx.x * 4 + wid;
  const bool valid = tile < ntTotal;
  if (!valid) tile = 0;            // idle waves compute tile 0, skip the store
  const int mt = tile / ntN, nt = tile % ntN;   // mt indexes 64-row blocks
  _Float16* buf0 = smem + wid * 5120;
  _Float16* buf1 = buf0 + 2560;
  const int lrow = lane >> 1, lseg = (lane & 1) * 16;
  unsigned l0A[4], l1A[4];
#pragma unroll
  for (int ti = 0; ti < 4; ++ti) {
    l0A[ti] = lds_off(buf0 + ti * 512 + lrow * 32 + lseg);
    l1A[ti] = lds_off(buf1 + ti * 512 + lrow * 32 + lseg);
  }
  const unsigned l0B = lds_off(buf0 + 2048 + lrow * 32 + lseg);
  const unsigned l1B = lds_off(buf1 + 2048 + lrow * 32 + lseg);
  const _Float16* a0 = A + (size_t)(mt * 64 + lrow) * lda + lseg;
  const _Float16* a1 = a0 + (size_t)16 * lda;
  const _Float16* a2 = a0 + (size_t)32 * lda;
  const _Float16* a3 = a0 + (size_t)48 * lda;
  const _Float16* bp = BT + (size_t)(nt * 16 + lrow) * ldbt + lseg;
  v8f acc[4] = {};
  const int nIter = K >> 5;        // even, >= 2
  stage_async10(l0A[0], l0A[1], l0A[2], l0A[3], l0B, a0, a1, a2, a3, bp);
  a0 += 32; a1 += 32; a2 += 32; a3 += 32; bp += 32;
  for (int it = 0; it + 2 < nIter; it += 2) {
    stage_async10(l1A[0], l1A[1], l1A[2], l1A[3], l1B, a0, a1, a2, a3, bp);
    a0 += 32; a1 += 32; a2 += 32; a3 += 32; bp += 32;
    wait_async_prev();
    compute4(buf0, acc, lane);
    stage_async10(l0A[0], l0A[1], l0A[2], l0A[3], l0B, a0, a1, a2, a3, bp);
    a0 += 32; a1 += 32; a2 += 32; a3 += 32; bp += 32;
    wait_async_prev();
    compute4(buf1, acc, lane);
  }
  stage_async10(l1A[0], l1A[1], l1A[2], l1A[3], l1B, a0, a1, a2, a3, bp);
  wait_async_prev();
  compute4(buf0, acc, lane);
  wait_async_all();
  compute4(buf1, acc, lane);
  if (valid) {
    const int hi = lane >> 4, nc = nt * 16 + (lane & 15);
#pragma unroll
    for (int ti = 0; ti < 4; ++ti)
#pragma unroll
      for (int r = 0; r < 8; ++r) {
        const int m = mt * 64 + ti * 16 + hi * 8 + r;
        if (m < M && nc < N) {
          const float v = acc[ti][r] + (bias ? bias[nc] : 0.f);
          if (Ch) Ch[(size_t)m * ldc + nc] = (_Float16)v;
          else    Cf[(size_t)m * ldc + nc] = v;
        }
      }
  }
}

// ---------------------------------------------------------------------------
// 3x3 SAME conv as implicit GEMM over a zero-padded f16 feature tensor
// (B x 10 x 34 x C), so the im2col A loader is branch-free and fully async.
// M=65536, N=512, K=4608 (144 K-chunks; each chunk sits in one (ky,kx) tap).
// ---------------------------------------------------------------------------
__global__ void convgemm_kernel(const _Float16* __restrict__ featPad,
                                const _Float16* __restrict__ WfT,
                                const float* __restrict__ bias,
                                _Float16* __restrict__ fpH) {
  __shared__ __align__(64) _Float16 smem[4 * 5120];
  const int wid = threadIdx.x >> 5, lane = threadIdx.x & 31;
  const int tile = blockIdx.x * 4 + wid;   // 32768 waves, exact multiple of 4
  const int mt = tile >> 5;                // / 32 N-tiles -> 64-row block
  const int nt = tile & 31;
  _Float16* buf0 = smem + wid * 5120;
  _Float16* buf1 = buf0 + 2560;
  const int lrow = lane >> 1, lseg = (lane & 1) * 16;
  unsigned l0A[4], l1A[4];
#pragma unroll
  for (int ti = 0; ti < 4; ++ti) {
    l0A[ti] = lds_off(buf0 + ti * 512 + lrow * 32 + lseg);
    l1A[ti] = lds_off(buf1 + ti * 512 + lrow * 32 + lseg);
  }
  const unsigned l0B = lds_off(buf0 + 2048 + lrow * 32 + lseg);
  const unsigned l1B = lds_off(buf1 + 2048 + lrow * 32 + lseg);
  // per-sub-tile pixel base in the padded tensor (tap (ky,kx) -> +ty,tx)
  const _Float16* pixB[4];
#pragma unroll
  for (int ti = 0; ti < 4; ++ti) {
    const int gm = mt * 64 + ti * 16 + lrow;
    const int b = gm >> 8, h = (gm >> 5) & 7, w = gm & 31;
    pixB[ti] = featPad + ((size_t)(b * kPH + h) * kPW + w) * kC + lseg;
  }
  const _Float16* bBase = WfT + (size_t)(nt * 16 + lrow) * kKc + lseg;
  auto stage = [&](int kc, const unsigned* lA, unsigned lB) {
    const int tap = kc >> 4;             // 16 chunks of 32 per tap (512 ch)
    const int ty = tap / 3, tx = tap % 3;
    const int d = ((ty * kPW + tx) << 9) + ((kc & 15) << 5);   // halves
    stage_async10(lA[0], lA[1], lA[2], lA[3], lB,
                  pixB[0] + d, pixB[1] + d, pixB[2] + d, pixB[3] + d,
                  bBase + (size_t)kc * 32);
  };
  v8f acc[4] = {};
  stage(0, l0A, l0B);
  for (int kc = 0; kc + 2 < 144; kc += 2) {
    stage(kc + 1, l1A, l1B);
    wait_async_prev();
    compute4(buf0, acc, lane);
    stage(kc + 2, l0A, l0B);
    wait_async_prev();
    compute4(buf1, acc, lane);
  }
  stage(143, l1A, l1B);
  wait_async_prev();
  compute4(buf0, acc, lane);
  wait_async_all();
  compute4(buf1, acc, lane);
  const int hi = lane >> 4, nc = nt * 16 + (lane & 15);
#pragma unroll
  for (int ti = 0; ti < 4; ++ti)
#pragma unroll
    for (int r = 0; r < 8; ++r) {
      const int m = mt * 64 + ti * 16 + hi * 8 + r;
      fpH[(size_t)m * kR + nc] = (_Float16)(acc[ti][r] + bias[nc]);
    }
}

// ------------------------- prep / elementwise kernels ----------------------
// zero-padded f16 copy of features: (B,8,32,C) f32 -> (B,10,34,C) f16 interior
__global__ void pad_feat_kernel(const float* __restrict__ src,
                                _Float16* __restrict__ dst) {
  const int i = blockIdx.x * blockDim.x + threadIdx.x;
  if (i >= kBHW * kC) return;
  const int c = i & 511;
  const int w = (i >> 9) & 31;
  const int h = (i >> 14) & 7;
  const int b = i >> 17;
  dst[((size_t)(b * kPH + h + 1) * kPW + (w + 1)) * kC + c] = (_Float16)src[i];
}

// D (cols x rows) = transpose of S (rows x cols), f32 -> f16
__global__ void transpose_to_f16_kernel(const float* __restrict__ S,
                                        _Float16* __restrict__ D,
                                        int rows, int cols) {
  const int i = blockIdx.x * blockDim.x + threadIdx.x;
  if (i >= rows * cols) return;
  const int n = i / rows, k = i % rows;    // D[n][k] = S[k][n]
  D[(size_t)n * rows + k] = (_Float16)S[(size_t)k * cols + n];
}

// WcatT (2048 x 1024) = [Wx; Wh]^T, f16
__global__ void build_wcatT_kernel(const float* __restrict__ Wx,
                                   const float* __restrict__ Wh,
                                   _Float16* __restrict__ WT) {
  const int i = blockIdx.x * blockDim.x + threadIdx.x;   // 2048*1024
  if (i >= 2048 * 1024) return;
  const int n = i >> 10, k = i & 1023;
  const float v = (k < kR) ? Wx[(size_t)k * 2048 + n]
                           : Wh[(size_t)(k - kR) * 2048 + n];
  WT[i] = (_Float16)v;
}

// WoutT (128 x 1024) = W_out^T padded with a zero row (n == 127)
__global__ void build_woutT_kernel(const float* __restrict__ Wout,
                                   _Float16* __restrict__ WT) {
  const int i = blockIdx.x * blockDim.x + threadIdx.x;   // 128*1024
  if (i >= 128 * 1024) return;
  const int n = i >> 10, k = i & 1023;
  WT[i] = (_Float16)((n < kV1) ? Wout[(size_t)k * kV1 + n] : 0.f);
}

// teacher-forced input: sym==V1 (start) -> zeros (out-of-range one_hot)
__global__ void build_x_kernel(const int* __restrict__ gt, int t,
                               const float* __restrict__ Wemb,
                               _Float16* __restrict__ xh0) {
  const int i = blockIdx.x * blockDim.x + threadIdx.x;   // 256*512
  if (i >= kB * kR) return;
  const int b = i >> 9, j = i & 511;
  const int sym = (t == 0) ? kV1 : gt[b * kT + (t - 1)];
  const float v = (sym >= 0 && sym < kV1) ? Wemb[(size_t)sym * kR + j] : 0.f;
  xh0[b * 1024 + j] = (_Float16)v;
}

__global__ void prime_x_kernel(const float* __restrict__ hol,
                               _Float16* __restrict__ xh0) {
  const int i = blockIdx.x * blockDim.x + threadIdx.x;   // 256*512
  if (i >= kB * kR) return;
  const int b = i >> 9, j = i & 511;
  xh0[b * 1024 + j] = (_Float16)hol[i];
}

__device__ __forceinline__ float sigmoidf_(float x) {
  return 1.f / (1.f + __expf(-x));
}

// z = (B,2048) gates [i,f,g,o]; c updated in place; h written as f16 copies
__global__ void lstm_act_kernel(const float* __restrict__ z,
                                float* __restrict__ c,
                                _Float16* d1, int ld1,
                                _Float16* d2, int ld2,
                                _Float16* d3, int ld3) {
  const int i = blockIdx.x * blockDim.x + threadIdx.x;   // 256*512
  if (i >= kB * kR) return;
  const int b = i >> 9, j = i & 511;
  const float* zb = z + (size_t)b * 2048;
  const float ig = zb[j], fg = zb[512 + j], gg = zb[1024 + j], og = zb[1536 + j];
  const float c2 = sigmoidf_(fg) * c[i] + sigmoidf_(ig) * tanhf(gg);
  const float h2 = sigmoidf_(og) * tanhf(c2);
  c[i] = c2;
  const _Float16 hh = (_Float16)h2;
  if (d1) d1[(size_t)b * ld1 + j] = hh;
  if (d2) d2[(size_t)b * ld2 + j] = hh;
  if (d3) d3[(size_t)b * ld3 + j] = hh;
}

// Bahdanau conv-attention + softmax + glimpse; fpH (67MB) + featPad (89MB)
// stay resident in the 192MB global L2 across all 31 steps.
__global__ void attn_kernel(const _Float16* __restrict__ fpH,
                            const float* __restrict__ hp,
                            const float* __restrict__ Wa,
                            const _Float16* __restrict__ featPad,
                            _Float16* __restrict__ catH) {
  const int b = blockIdx.x, tid = threadIdx.x;   // 256 threads
  __shared__ float hs[512];
  __shared__ float wa[512];
  __shared__ float sc[256];
  __shared__ float red[256];
  __shared__ float bc[2];
  hs[tid] = hp[b * kR + tid];
  hs[tid + 256] = hp[b * kR + tid + 256];
  wa[tid] = Wa[tid];
  wa[tid + 256] = Wa[tid + 256];
  __syncthreads();
  // score for position p = tid: 64 x 16B vector reads of fp
  const v8h* f8 = (const v8h*)(fpH + ((size_t)b * kHW + tid) * kR);
  float s = 0.f;
  for (int a8 = 0; a8 < kR / 8; ++a8) {
    const v8h v = f8[a8];
    const int base = a8 * 8;
#pragma unroll
    for (int j = 0; j < 8; ++j)
      s += tanhf((float)v[j] + hs[base + j]) * wa[base + j];
  }
  red[tid] = s;
  __syncthreads();
  for (int st = 128; st > 0; st >>= 1) {
    if (tid < st) red[tid] = fmaxf(red[tid], red[tid + st]);
    __syncthreads();
  }
  if (tid == 0) bc[0] = red[0];
  __syncthreads();
  const float e = __expf(s - bc[0]);
  red[tid] = e;
  __syncthreads();
  for (int st = 128; st > 0; st >>= 1) {
    if (tid < st) red[tid] += red[tid + st];
    __syncthreads();
  }
  if (tid == 0) bc[1] = red[0];
  __syncthreads();
  sc[tid] = e / bc[1];          // softmax weight for position tid
  __syncthreads();
  // glimpse: 2 adjacent channels per thread -> 4B/lane coalesced reads
  const _Float16* fb =
      featPad + ((size_t)(b * kPH + 1) * kPW + 1) * kC + 2 * tid;
  float g0 = 0.f, g1 = 0.f;
  for (int p = 0; p < kHW; ++p) {
    const float wp = sc[p];
    const size_t poff = (size_t)((p >> 5) * kPW + (p & 31)) * kC;
    const v2h v = *(const v2h*)(fb + poff);
    g0 += wp * (float)v[0];
    g1 += wp * (float)v[1];
  }
  catH[(size_t)b * 1024 + 512 + 2 * tid]     = (_Float16)g0;
  catH[(size_t)b * 1024 + 512 + 2 * tid + 1] = (_Float16)g1;
}

// ---------------------------------------------------------------------------
extern "C" void kernel_launch(void* const* d_in, const int* in_sizes, int n_in,
                              void* d_out, int out_size, void* d_ws, size_t ws_size,
                              hipStream_t stream) {
  const float* features = (const float*)d_in[0];
  const float* holistic = (const float*)d_in[1];
  const int*   gt       = (const int*)d_in[2];
  const float* W_embed  = (const float*)d_in[3];
  const float* Wx0      = (const float*)d_in[4];
  const float* Wh0      = (const float*)d_in[5];
  const float* b0       = (const float*)d_in[6];
  const float* Wx1      = (const float*)d_in[7];
  const float* Wh1      = (const float*)d_in[8];
  const float* b1       = (const float*)d_in[9];
  const float* Wh_att   = (const float*)d_in[10];
  const float* Wf_att   = (const float*)d_in[11];
  const float* bf_att   = (const float*)d_in[12];
  const float* Wa_att   = (const float*)d_in[13];
  const float* W_out    = (const float*)d_in[14];
  const float* b_out    = (const float*)d_in[15];
  float* out = (float*)d_out;

  // ---- workspace carve-up (~175 MB) ----
  char* wsb = (char*)d_ws;
  size_t off = 0;
  auto alloc = [&](size_t bytes) -> void* {
    void* p = wsb + off;
    off = (off + bytes + 255) & ~(size_t)255;
    return p;
  };
  const size_t featPadBytes = (size_t)kB * kPH * kPW * kC * 2;      // 89 MB
  _Float16* featPad = (_Float16*)alloc(featPadBytes);
  _Float16* fpH    = (_Float16*)alloc((size_t)kBHW * kR * 2);       // 67 MB
  _Float16* WfT    = (_Float16*)alloc((size_t)kR * kKc * 2);        // 512x4608
  _Float16* Wcat0T = (_Float16*)alloc((size_t)2048 * 1024 * 2);
  _Float16* Wcat1T = (_Float16*)alloc((size_t)2048 * 1024 * 2);
  _Float16* WhaT   = (_Float16*)alloc((size_t)kR * kR * 2);
  _Float16* WoutT  = (_Float16*)alloc((size_t)128 * 1024 * 2);
  float*    z      = (float*)alloc((size_t)kB * 2048 * 4);
  float*    c0     = (float*)alloc((size_t)kB * kR * 4);
  float*    c1     = (float*)alloc((size_t)kB * kR * 4);
  float*    hp     = (float*)alloc((size_t)kB * kR * 4);
  _Float16* xh0    = (_Float16*)alloc((size_t)kB * 1024 * 2);
  _Float16* xh1    = (_Float16*)alloc((size_t)kB * 1024 * 2);
  _Float16* catH   = (_Float16*)alloc((size_t)kB * 1024 * 2);
  _Float16* h1H    = (_Float16*)alloc((size_t)kB * kR * 2);

  const dim3 blk(256);
  auto nblk = [](int n) { return dim3((unsigned)((n + 255) / 256)); };

  // ---- one-time (per call) prep: conversions + zero state ----
  hipMemsetAsync(featPad, 0, featPadBytes, stream);
  hipMemsetAsync(c0, 0, (size_t)kB * kR * 4, stream);
  hipMemsetAsync(c1, 0, (size_t)kB * kR * 4, stream);
  hipMemsetAsync(xh0, 0, (size_t)kB * 1024 * 2, stream);
  hipMemsetAsync(xh1, 0, (size_t)kB * 1024 * 2, stream);

  pad_feat_kernel<<<nblk(kBHW * kC), blk, 0, stream>>>(features, featPad);
  transpose_to_f16_kernel<<<nblk(kKc * kR), blk, 0, stream>>>(Wf_att, WfT,
                                                              kKc, kR);
  transpose_to_f16_kernel<<<nblk(kR * kR), blk, 0, stream>>>(Wh_att, WhaT,
                                                             kR, kR);
  build_wcatT_kernel<<<nblk(2048 * 1024), blk, 0, stream>>>(Wx0, Wh0, Wcat0T);
  build_wcatT_kernel<<<nblk(2048 * 1024), blk, 0, stream>>>(Wx1, Wh1, Wcat1T);
  build_woutT_kernel<<<nblk(128 * 1024), blk, 0, stream>>>(W_out, WoutT);

  // ---- conv projection once: fpH = conv3x3(features) + bf_att (f16) ----
  convgemm_kernel<<<dim3(32768 / 4), dim3(128), 0, stream>>>(featPad, WfT,
                                                             bf_att, fpH);

  // GEMM launch helper (BT = transposed weights, ldbt = K); M % 64 == 0
  auto gemm = [&](const _Float16* A, int lda, const _Float16* BT, int ldbt,
                  const float* bias, float* Cf, _Float16* Ch, int ldc,
                  int M, int N, int K) {
    const int ntN = (N + 15) / 16;
    const int waves = (M >> 6) * ntN;
    gemm_wmma_kernel<<<dim3((unsigned)((waves + 3) / 4)), dim3(128), 0,
                       stream>>>(A, lda, BT, ldbt, bias, Cf, Ch, ldc, M, N, K,
                                 ntN, waves);
  };

  // ---- prime the LSTM stack with the holistic feature ----
  prime_x_kernel<<<nblk(kB * kR), blk, 0, stream>>>(holistic, xh0);
  gemm(xh0, 1024, Wcat0T, 1024, b0, z, nullptr, 2048, kB, 2048, 1024);
  lstm_act_kernel<<<nblk(kB * kR), blk, 0, stream>>>(
      z, c0, xh1, 1024, xh0 + 512, 1024, nullptr, 0);
  gemm(xh1, 1024, Wcat1T, 1024, b1, z, nullptr, 2048, kB, 2048, 1024);
  lstm_act_kernel<<<nblk(kB * kR), blk, 0, stream>>>(
      z, c1, xh1 + 512, 1024, catH, 1024, h1H, kR);

  // ---- 31 teacher-forced decode steps ----
  for (int t = 0; t < kT; ++t) {
    build_x_kernel<<<nblk(kB * kR), blk, 0, stream>>>(gt, t, W_embed, xh0);
    gemm(xh0, 1024, Wcat0T, 1024, b0, z, nullptr, 2048, kB, 2048, 1024);
    lstm_act_kernel<<<nblk(kB * kR), blk, 0, stream>>>(
        z, c0, xh1, 1024, xh0 + 512, 1024, nullptr, 0);
    gemm(xh1, 1024, Wcat1T, 1024, b1, z, nullptr, 2048, kB, 2048, 1024);
    lstm_act_kernel<<<nblk(kB * kR), blk, 0, stream>>>(
        z, c1, xh1 + 512, 1024, catH, 1024, h1H, kR);
    // hp = h1 @ Wh_att
    gemm(h1H, kR, WhaT, kR, nullptr, hp, nullptr, kR, kB, kR, kR);
    // attention + glimpse -> catH[:, 512:]
    attn_kernel<<<dim3(kB), blk, 0, stream>>>(fpH, hp, Wa_att, featPad, catH);
    // logits_t = [h1, glimpse] @ W_out + b_out  -> out[:, t, :]
    gemm(catH, 1024, WoutT, 1024, b_out, out + (size_t)t * kV1, nullptr,
         kT * kV1, kB, kV1, 1024);
  }
}
